// UniPhyBlock_71820443123825
// MI455X (gfx1250) — compile-verified, matrix operand order checked by reading, not run on hardware
//
#include <hip/hip_runtime.h>
#include <hip/hip_bf16.h>

// ---------------------------------------------------------------------------
// UniPhyBlock for MI455X (gfx1250, wave32).  All dense contractions run on
// v_wmma_f32_16x16x32_f16 (f32 accumulate).  Weights are pre-packed into the
// exact CDNA5 B-fragment VGPR layout so each fragment is one 32B load.
// Conv3x3 stages a zero-padded input patch in LDS (shared by the 4 waves of
// the block) so every A fragment is two contiguous ds_load_b128, branchless.
// Bulk state stays f32 (bandwidth-bound problem, ~1GB traffic @ 23.3TB/s).
//
// Sizes: B=2 T=24 D=32 H=W=64  ->  NPIX=196608, N32=6291456.
// ws usage: 4*N32 f32 + ~0.41M f32 + ~60K f16  ~= 103 MB.
// ---------------------------------------------------------------------------

typedef __attribute__((ext_vector_type(16))) _Float16 v16h;
typedef __attribute__((ext_vector_type(8)))  float    v8f;

static constexpr int    CB   = 2;
static constexpr int    CT   = 24;
static constexpr int    CD   = 32;
static constexpr int    CHW  = 4096;       // 64*64
static constexpr int    NPIX = 196608;     // B*T*H*W
static constexpr size_t N32  = 6291456;    // NPIX * D

__device__ __forceinline__ v8f wmma16(v16h a, v16h b, v8f c) {
  return __builtin_amdgcn_wmma_f32_16x16x32_f16(
      /*neg_a=*/false, a, /*neg_b=*/false, b,
      /*c_mod=*/(short)0, c, /*reuse_a=*/false, /*reuse_b=*/false);
}

// A fragment from row-major f32; p = base + row*ld + kbase + koff
__device__ __forceinline__ v16h a_frag_rm(const float* p) {
  v16h a;
#pragma unroll
  for (int j = 0; j < 8; ++j) a[j] = (_Float16)p[j];
#pragma unroll
  for (int j = 0; j < 8; ++j) a[8 + j] = (_Float16)p[16 + j];
  return a;
}

// A fragment from strided (channel-major) f32: element k at base[k*stride]
__device__ __forceinline__ v16h a_frag_str(const float* base, int koff, int stride) {
  v16h a;
#pragma unroll
  for (int j = 0; j < 8; ++j) a[j] = (_Float16)base[(koff + j) * stride];
#pragma unroll
  for (int j = 0; j < 8; ++j) a[8 + j] = (_Float16)base[(koff + 16 + j) * stride];
  return a;
}

// Packed B fragment: [frag][lane][16] halves, 32B aligned
__device__ __forceinline__ v16h b_frag(const _Float16* pk, int frag, int lane) {
  return *(const v16h*)(pk + (size_t)frag * 512 + lane * 16);
}

__device__ __forceinline__ float softplus_f(float x) {
  return (x > 20.f) ? x : log1pf(__expf(x));
}
__device__ __forceinline__ float sigmoid_f(float x) { return 1.f / (1.f + __expf(-x)); }
__device__ __forceinline__ float silu_f(float x)    { return x / (1.f + __expf(-x)); }

// ------------------------- small setup kernels -----------------------------

// complex 32x32 Gauss-Jordan inverse (one wave)
__global__ void k_inv(const float* Wre, const float* Wim, float* Vre, float* Vim) {
  __shared__ float Ar[32][64], Ai[32][64];
  int r = threadIdx.x;
  for (int c = 0; c < 32; ++c) { Ar[r][c] = Wre[r * 32 + c]; Ai[r][c] = Wim[r * 32 + c]; }
  for (int c = 0; c < 32; ++c) { Ar[r][32 + c] = (c == r) ? 1.f : 0.f; Ai[r][32 + c] = 0.f; }
  __syncthreads();
  for (int k = 0; k < 32; ++k) {
    if (r == k) {
      float pr = Ar[k][k], pi = Ai[k][k];
      float den = pr * pr + pi * pi;
      float ir = pr / den, ii = -pi / den;
      for (int c = 0; c < 64; ++c) {
        float ar = Ar[k][c], ai = Ai[k][c];
        Ar[k][c] = ar * ir - ai * ii; Ai[k][c] = ar * ii + ai * ir;
      }
    }
    __syncthreads();
    if (r != k) {
      float fr = Ar[r][k], fi = Ai[r][k];
      for (int c = 0; c < 64; ++c) {
        float ar = Ar[k][c], ai = Ai[k][c];
        Ar[r][c] -= fr * ar - fi * ai;
        Ai[r][c] -= fr * ai + fi * ar;
      }
    }
    __syncthreads();
  }
  for (int c = 0; c < 32; ++c) { Vre[r * 32 + c] = Ar[r][32 + c]; Vim[r * 32 + c] = Ai[r][32 + c]; }
}

// pack row-major (or transposed) f32 [K][N] weight into B-fragment layout
__global__ void k_pack_rm(const float* src, _Float16* dst, int K, int N, int transposed) {
  int tid = blockIdx.x * blockDim.x + threadIdx.x;
  int cgs = N >> 4;
  int nthr = (K >> 5) * cgs * 32;
  if (tid >= nthr) return;
  int lane = tid & 31, frag = tid >> 5;
  int chunk = frag / cgs, cg = frag % cgs;
  int col = cg * 16 + (lane & 15);
  int koff = (lane >> 4) << 3;
#pragma unroll
  for (int j = 0; j < 16; ++j) {
    int kk = chunk * 32 + koff + (j < 8 ? j : 8 + j);
    float v = transposed ? src[col * K + kk] : src[kk * N + col];
    dst[(size_t)frag * 512 + lane * 16 + j] = (_Float16)v;
  }
}

// pack OIHW 3x3 conv weights; K index = (ky*3+kx)*64 + ci, col = co
__global__ void k_pack_conv(const float* w, _Float16* dst) {
  int tid = blockIdx.x * blockDim.x + threadIdx.x;       // 18*4*32 = 2304
  if (tid >= 2304) return;
  int lane = tid & 31, frag = tid >> 5;
  int chunk = frag >> 2, cg = frag & 3;
  int co = cg * 16 + (lane & 15);
  int koff = (lane >> 4) << 3;
#pragma unroll
  for (int j = 0; j < 16; ++j) {
    int kk = chunk * 32 + koff + (j < 8 ? j : 8 + j);
    int tap = kk >> 6, ci = kk & 63;
    int ky = tap / 3, kx = tap - ky * 3;
    dst[(size_t)frag * 512 + lane * 16 + j] =
        (_Float16)w[((co * 64 + ci) * 3 + ky) * 3 + kx];
  }
}

// ------------------------- stage 1: spatial LN + conv ----------------------

// per-pixel LN over 64 channels (re||im), write xn channel-major [BT][64][HW]
__global__ void k_ln_spatial(const float* xre, const float* xim,
                             const float* g, const float* b, float* xn) {
  int p = blockIdx.x * blockDim.x + threadIdx.x;
  if (p >= NPIX) return;
  int f = p >> 12, pf = p & 4095;
  float v[64], s = 0.f, s2 = 0.f;
#pragma unroll
  for (int c = 0; c < 64; ++c) {
    float x = (c < 32) ? xre[((size_t)f * 32 + c) * CHW + pf]
                       : xim[((size_t)f * 32 + c - 32) * CHW + pf];
    v[c] = x; s += x; s2 += x * x;
  }
  float m = s * (1.f / 64.f);
  float var = s2 * (1.f / 64.f) - m * m;
  float rs = rsqrtf(var + 1e-5f);
#pragma unroll
  for (int c = 0; c < 64; ++c)
    xn[((size_t)f * 64 + c) * CHW + pf] = (v[c] - m) * rs * g[c] + b[c];
}

// implicit-GEMM 3x3 conv 64->64 + bias + residual; writes x_post into d_out.
// One block = one 16-pixel strip; 4 waves cover the 4 out-channel groups and
// share a zero-padded f16 input patch in LDS ([ky][x][ci], ci padded to 72
// halves so the 16 A-lanes hit disjoint bank groups).  A K-chunk of 32 never
// crosses a filter tap, so each A fragment is two contiguous b128 LDS loads.
__global__ void __launch_bounds__(128) k_conv(const float* xn, const _Float16* pk,
                                              const float* bias, const float* xre,
                                              const float* xim, float* out) {
  __shared__ _Float16 patch[3][18][72];
  int tid = threadIdx.x;
  int lane = tid & 31;
  int cg = tid >> 5;
  int s = blockIdx.x;
  int f = s >> 8;
  int pti = s & 255;
  int y = pti >> 2;
  int x0 = (pti & 3) << 4;

  // cooperative, branchless staging (clamped address + select-zero)
  const float* base = xn + (size_t)f * 64 * CHW;
  for (int e = tid; e < 3456; e += 128) {
    int ci = e & 63;
    int r = e >> 6;               // 0..53
    int xx = r % 18;
    int yy = r / 18;
    int gy = y + yy - 1, gx = x0 + xx - 1;
    bool ok = (gy >= 0) && (gy < 64) && (gx >= 0) && (gx < 64);
    int cy = ok ? gy : 0, cx = ok ? gx : 0;
    float v = base[((size_t)ci << 12) + (cy << 6) + cx];
    patch[yy][xx][ci] = (_Float16)(ok ? v : 0.f);
  }
  __syncthreads();

  int col = cg * 16 + (lane & 15);
  int koff = (lane >> 4) << 3;
  float bv = bias[col];
  v8f acc;
#pragma unroll
  for (int v = 0; v < 8; ++v) acc[v] = bv;
  for (int ch = 0; ch < 18; ++ch) {
    int tap = ch >> 1;
    int ky = tap / 3, kx = tap - ky * 3;
    int cb = (ch & 1) << 5;
    const _Float16* pp = &patch[ky][(lane & 15) + kx][cb + koff];
    v16h a;
#pragma unroll
    for (int j = 0; j < 8; ++j) a[j] = pp[j];
#pragma unroll
    for (int j = 0; j < 8; ++j) a[8 + j] = pp[16 + j];
    acc = wmma16(a, b_frag(pk, ch * 4 + cg, lane), acc);
  }
  int rowofs = (lane >> 4) << 3;
  int d = col & 31;
  const float* xin = (col < 32) ? xre : xim;
  float* dst = out + (col < 32 ? (size_t)0 : N32);
#pragma unroll
  for (int v = 0; v < 8; ++v) {
    int px = x0 + rowofs + v;
    size_t idx = ((size_t)f * 32 + d) * CHW + (y << 6) + px;
    dst[idx] = xin[idx] + acc[v];
  }
}

// ------------------------- stage 2: complex encode -------------------------

__global__ void __launch_bounds__(128) k_encode(const float* xp, const _Float16* pbre,
                                                const _Float16* pbim, float* xer, float* xei) {
  int lane = threadIdx.x & 31;
  int tile = (blockIdx.x << 2) + (threadIdx.x >> 5);
  int g0 = tile << 4;
  int gA = g0 + (lane & 15);
  int f = gA >> 12, pf = gA & 4095;
  int koff = (lane >> 4) << 3;
  const float* bre = xp + (size_t)f * 32 * CHW + pf;
  const float* bim = xp + N32 + (size_t)f * 32 * CHW + pf;
  v16h are = a_frag_str(bre, koff, CHW);
  v16h aim = a_frag_str(bim, koff, CHW);
  int rowofs = (lane >> 4) << 3;
#pragma unroll
  for (int cg = 0; cg < 2; ++cg) {
    v8f rr = {}; v8f ii = {}; v8f ri = {}; v8f ir = {};
    v16h br = b_frag(pbre, cg, lane);
    v16h bi = b_frag(pbim, cg, lane);
    rr = wmma16(are, br, rr);
    ii = wmma16(aim, bi, ii);
    ri = wmma16(are, bi, ri);
    ir = wmma16(aim, br, ir);
    int col = cg * 16 + (lane & 15);
#pragma unroll
    for (int v = 0; v < 8; ++v) {
      size_t row = (size_t)(g0 + rowofs + v);
      xer[row * 32 + col] = rr[v] - ii[v];
      xei[row * 32 + col] = ri[v] + ir[v];
    }
  }
}

// ------------------------- flux path ---------------------------------------

__global__ void k_xmean(const float* xer, const float* xei, float* xmr, float* xmi) {
  __shared__ float sr[256], si[256];
  int btd = blockIdx.x;                // 0..1535
  int btf = btd >> 5, d = btd & 31;
  int t = threadIdx.x;
  float ar = 0.f, ai = 0.f;
  for (int p = t; p < CHW; p += 256) {
    size_t idx = ((size_t)btf * CHW + p) * 32 + d;
    ar += xer[idx]; ai += xei[idx];
  }
  sr[t] = ar; si[t] = ai; __syncthreads();
  for (int s = 128; s > 0; s >>= 1) {
    if (t < s) { sr[t] += sr[t + s]; si[t] += si[t + s]; }
    __syncthreads();
  }
  if (t == 0) { xmr[btd] = sr[0] * (1.f / 4096.f); xmi[btd] = si[0] * (1.f / 4096.f); }
}

__global__ void k_flux(const float* dt_seq, const float* lam_flux,
                       const float* bfre, const float* bfim,
                       const float* fre0, const float* fim0,
                       const float* cre, const float* cim,
                       const float* gw, const float* gb,
                       const float* adec, const float* bfrq,
                       const float* xmr, const float* xmi,
                       float* gate, float* srcr, float* srci,
                       float* odr, float* odi, float* ofr, float* ofi,
                       float* outflux) {
  int tid = threadIdx.x;               // 64
  int b = tid >> 5, d = tid & 31;
  float sp = softplus_f(lam_flux[d]);
  float fr = fre0[b * 32 + d], fi = fim0[b * 32 + d];
  float lamr = -softplus_f(adec[d]), lami = bfrq[d];
  float den = lamr * lamr + lami * lami;
  for (int t = 0; t < CT; ++t) {
    int idx = (b * CT + t) * 32 + d;
    float dt = dt_seq[b * CT + t];
    float A = __expf(-sp * dt);
    float xr = xmr[idx], xi = xmi[idx];
    float Xr = (bfre[d] * xr - bfim[d] * xi) * dt;
    float Xi = (bfre[d] * xi + bfim[d] * xr) * dt;
    fr = A * fr + Xr; fi = A * fi + Xi;
    gate[idx] = sigmoid_f(fr * gw[d] + gb[d]);
    srcr[idx] = fr * cre[d] - fi * cim[d];
    srci[idx] = fr * cim[d] + fi * cre[d];
    float er = __expf(lamr * dt);
    float odre = er * __cosf(lami * dt), odim = er * __sinf(lami * dt);
    odr[idx] = odre; odi[idx] = odim;
    ofr[idx] = ((odre - 1.f) * lamr + odim * lami) / den;
    ofi[idx] = (odim * lamr - (odre - 1.f) * lami) / den;
  }
  outflux[b * 32 + d] = fr;
  outflux[64 + b * 32 + d] = fi;
}

// ------------------------- spatial gate chain ------------------------------

__global__ void __launch_bounds__(128) k_pw1(const float* xer, const float* xei,
                                             const _Float16* pk, const float* bias, float* g1) {
  int lane = threadIdx.x & 31;
  int tile = (blockIdx.x << 2) + (threadIdx.x >> 5);
  int g0 = tile << 4;
  size_t row = (size_t)(g0 + (lane & 15));
  int koff = (lane >> 4) << 3;
  v16h a0 = a_frag_rm(xer + row * 32 + koff);
  v16h a1 = a_frag_rm(xei + row * 32 + koff);
  int rowofs = (lane >> 4) << 3;
#pragma unroll
  for (int cg = 0; cg < 2; ++cg) {
    int col = cg * 16 + (lane & 15);
    float bv = bias[col];
    v8f acc;
#pragma unroll
    for (int v = 0; v < 8; ++v) acc[v] = bv;
    acc = wmma16(a0, b_frag(pk, cg, lane), acc);
    acc = wmma16(a1, b_frag(pk, 2 + cg, lane), acc);
#pragma unroll
    for (int v = 0; v < 8; ++v)
      g1[(size_t)(g0 + rowofs + v) * 32 + col] = silu_f(acc[v]);
  }
}

__global__ void k_dw(const float* g1, const float* w, float* g2) {
  size_t tid = (size_t)blockIdx.x * blockDim.x + threadIdx.x;
  if (tid >= N32) return;
  int d = (int)(tid & 31);
  int p = (int)((tid >> 5) & 4095);
  int f = (int)(tid >> 17);
  int y = p >> 6, x = p & 63;
  float s = 0.f;
#pragma unroll
  for (int ky = 0; ky < 3; ++ky)
#pragma unroll
    for (int kx = 0; kx < 3; ++kx) {
      int yy = y + ky - 1, xx = x + kx - 1;
      bool ok = (yy >= 0) && (yy < 64) && (xx >= 0) && (xx < 64);
      int cy = ok ? yy : 0, cx = ok ? xx : 0;
      float v = g1[(((size_t)f * CHW) + (cy << 6) + cx) * 32 + d];
      s += (ok ? v : 0.f) * w[d * 9 + ky * 3 + kx];
    }
  g2[tid] = s;
}

__global__ void __launch_bounds__(128) k_pw2(const float* g2, const _Float16* pk,
                                             const float* bias, float* spg) {
  int lane = threadIdx.x & 31;
  int tile = (blockIdx.x << 2) + (threadIdx.x >> 5);
  int g0 = tile << 4;
  size_t row = (size_t)(g0 + (lane & 15));
  int koff = (lane >> 4) << 3;
  v16h a = a_frag_rm(g2 + row * 32 + koff);
  int rowofs = (lane >> 4) << 3;
#pragma unroll
  for (int cg = 0; cg < 2; ++cg) {
    int col = cg * 16 + (lane & 15);
    float bv = bias[col];
    v8f acc;
#pragma unroll
    for (int v = 0; v < 8; ++v) acc[v] = bv;
    acc = wmma16(a, b_frag(pk, cg, lane), acc);
#pragma unroll
    for (int v = 0; v < 8; ++v)
      spg[(size_t)(g0 + rowofs + v) * 32 + col] = sigmoid_f(acc[v]);
  }
}

// ------------------------- temporal scan -----------------------------------

__global__ void k_scan(float* xer, float* xei, const float* spg,
                       const float* gate, const float* srcr, const float* srci,
                       const float* odr, const float* odi,
                       const float* ofr, const float* ofi,
                       const float* hre, const float* him, float* outh) {
  int tid = blockIdx.x * blockDim.x + threadIdx.x;    // 262144
  int d = tid & 31;
  int p = (tid >> 5) & 4095;
  int b = tid >> 17;
  const size_t tstride = (size_t)CHW * 32;
  float ur = 0.f, ui = 0.f;
  for (int t = 0; t < CT; ++t) {
    size_t idx = (((size_t)(b * CT + t)) * CHW + p) * 32 + d;
    if (t + 1 < CT) {   // hide next-step latency (global_prefetch_b8)
      __builtin_prefetch(&xer[idx + tstride], 0, 1);
      __builtin_prefetch(&xei[idx + tstride], 0, 1);
      __builtin_prefetch(&spg[idx + tstride], 0, 1);
    }
    int s = (b * CT + t) * 32 + d;
    float xr = xer[idx], xi = xei[idx];
    float gc = gate[s] * spg[idx];
    float fr = xr * gc + srcr[s] * (1.f - gc);
    float fi = xi * gc + srci[s] * (1.f - gc);
    float o_r = ofr[s], o_i = ofi[s];
    float utr = fr * o_r - fi * o_i;
    float uti = fr * o_i + fi * o_r;
    float ar = odr[s], ai = odi[s];
    if (t == 0) {
      size_t hi0 = ((size_t)b * CHW + p) * 32 + d;
      float hr = hre[hi0], hh = him[hi0];
      utr += hr * ar - hh * ai;
      uti += hr * ai + hh * ar;
    }
    float nr = ar * ur - ai * ui + utr;
    float ni = ar * ui + ai * ur + uti;
    ur = nr; ui = ni;
    xer[idx] = ur; xei[idx] = ui;
  }
  size_t hidx = ((size_t)b * CHW + p) * 32 + d;
  outh[hidx] = ur;
  outh[262144 + hidx] = ui;
}

// ------------------------- decode + LN stats + FFN -------------------------

__global__ void __launch_bounds__(128) k_decode(const float* xer, const float* xei,
                                                const _Float16* pbre, const _Float16* pbim,
                                                float* dre, float* dim) {
  int lane = threadIdx.x & 31;
  int tile = (blockIdx.x << 2) + (threadIdx.x >> 5);
  int g0 = tile << 4;
  size_t row = (size_t)(g0 + (lane & 15));
  int koff = (lane >> 4) << 3;
  v16h are = a_frag_rm(xer + row * 32 + koff);
  v16h aim = a_frag_rm(xei + row * 32 + koff);
  int rowofs = (lane >> 4) << 3;
#pragma unroll
  for (int cg = 0; cg < 2; ++cg) {
    v8f rr = {}; v8f ii = {}; v8f ri = {}; v8f ir = {};
    v16h br = b_frag(pbre, cg, lane);
    v16h bi = b_frag(pbim, cg, lane);
    rr = wmma16(are, br, rr);
    ii = wmma16(aim, bi, ii);
    ri = wmma16(are, bi, ri);
    ir = wmma16(aim, br, ir);
    int col = cg * 16 + (lane & 15);
#pragma unroll
    for (int v = 0; v < 8; ++v) {
      size_t r2 = (size_t)(g0 + rowofs + v);
      dre[r2 * 32 + col] = rr[v] - ii[v];
      dim[r2 * 32 + col] = ri[v] + ir[v];
    }
  }
}

__global__ void k_lnstats(const float* dre, const float* dim, float* mean, float* rstd) {
  int p = blockIdx.x * blockDim.x + threadIdx.x;
  if (p >= NPIX) return;
  float s = 0.f, s2 = 0.f;
#pragma unroll
  for (int c = 0; c < 32; ++c) { float a = dre[(size_t)p * 32 + c]; s += a; s2 += a * a; }
#pragma unroll
  for (int c = 0; c < 32; ++c) { float a = dim[(size_t)p * 32 + c]; s += a; s2 += a * a; }
  float m = s * (1.f / 64.f);
  float var = s2 * (1.f / 64.f) - m * m;
  mean[p] = m;
  rstd[p] = rsqrtf(var + 1e-5f);
}

// fused LN -> FFN(64->128 SiLU ->64) -> delta; z = x_post + dec + delta into d_out
__global__ void __launch_bounds__(128) k_ffn(const float* dre, const float* dim,
                                             const float* mean, const float* rstd,
                                             const float* lng, const float* lnb,
                                             const _Float16* pk1, const float* b1,
                                             const _Float16* pk2, const float* b2,
                                             float* out) {
  __shared__ _Float16 hid[4][16][128];
  int lane = threadIdx.x & 31;
  int wv = threadIdx.x >> 5;
  int tile = (blockIdx.x << 2) + wv;
  int g0 = tile << 4;
  size_t rowA = (size_t)(g0 + (lane & 15));
  int koff = (lane >> 4) << 3;
  float m = mean[rowA], rs = rstd[rowA];
  v16h aA, aB;
#pragma unroll
  for (int j = 0; j < 16; ++j) {
    int k = koff + (j < 8 ? j : 8 + j);
    float v0 = dre[rowA * 32 + k];
    aA[j] = (_Float16)((v0 - m) * rs * lng[k] + lnb[k]);
    float v1 = dim[rowA * 32 + k];
    aB[j] = (_Float16)((v1 - m) * rs * lng[32 + k] + lnb[32 + k]);
  }
  int rowofs = (lane >> 4) << 3;
  int cl = lane & 15;
  // GEMM1: [16x64] x [64x128], SiLU -> LDS
#pragma unroll
  for (int cg = 0; cg < 8; ++cg) {
    int col = cg * 16 + cl;
    float bv = b1[col];
    v8f acc;
#pragma unroll
    for (int v = 0; v < 8; ++v) acc[v] = bv;
    acc = wmma16(aA, b_frag(pk1, cg, lane), acc);
    acc = wmma16(aB, b_frag(pk1, 8 + cg, lane), acc);
#pragma unroll
    for (int v = 0; v < 8; ++v)
      hid[wv][rowofs + v][col] = (_Float16)silu_f(acc[v]);
  }
  // GEMM2: [16x128] x [128x64] + final residual store
  int f = g0 >> 12;
#pragma unroll
  for (int cg = 0; cg < 4; ++cg) {
    int col = cg * 16 + cl;
    float bv = b2[col];
    v8f acc;
#pragma unroll
    for (int v = 0; v < 8; ++v) acc[v] = bv;
#pragma unroll
    for (int ch = 0; ch < 4; ++ch) {
      const _Float16* hp = &hid[wv][cl][ch * 32 + koff];
      v16h a;
#pragma unroll
      for (int j = 0; j < 8; ++j) a[j] = hp[j];
#pragma unroll
      for (int j = 0; j < 8; ++j) a[8 + j] = hp[16 + j];
      acc = wmma16(a, b_frag(pk2, ch * 4 + cg, lane), acc);
    }
#pragma unroll
    for (int v = 0; v < 8; ++v) {
      size_t row = (size_t)(g0 + rowofs + v);
      int pf = (int)(row & 4095);
      float delta = acc[v];
      if (col < 32) {
        size_t oi = ((size_t)f * 32 + col) * CHW + pf;
        out[oi] = out[oi] + dre[row * 32 + col] + delta;
      } else {
        int dd = col - 32;
        size_t oi = N32 + ((size_t)f * 32 + dd) * CHW + pf;
        out[oi] = out[oi] + dim[row * 32 + dd] + delta;
      }
    }
  }
}

// ---------------------------------------------------------------------------

extern "C" void kernel_launch(void* const* d_in, const int* in_sizes, int n_in,
                              void* d_out, int out_size, void* d_ws, size_t ws_size,
                              hipStream_t stream) {
  const float* x_re     = (const float*)d_in[0];
  const float* x_im     = (const float*)d_in[1];
  const float* h_re     = (const float*)d_in[2];
  const float* h_im     = (const float*)d_in[3];
  const float* flux_re  = (const float*)d_in[4];
  const float* flux_im  = (const float*)d_in[5];
  const float* dt_seq   = (const float*)d_in[6];
  const float* ln_s_g   = (const float*)d_in[7];
  const float* ln_s_b   = (const float*)d_in[8];
  const float* conv_w   = (const float*)d_in[9];
  const float* conv_b   = (const float*)d_in[10];
  const float* ln_t_g   = (const float*)d_in[11];
  const float* ln_t_b   = (const float*)d_in[12];
  const float* W_re     = (const float*)d_in[13];
  const float* W_im     = (const float*)d_in[14];
  const float* lam_flux = (const float*)d_in[15];
  const float* bf_re    = (const float*)d_in[16];
  const float* bf_im    = (const float*)d_in[17];
  const float* c_re     = (const float*)d_in[18];
  const float* c_im     = (const float*)d_in[19];
  const float* g_w      = (const float*)d_in[20];
  const float* g_b      = (const float*)d_in[21];
  const float* a_decay  = (const float*)d_in[22];
  const float* b_freq   = (const float*)d_in[23];
  const float* ffn_w1   = (const float*)d_in[24];
  const float* ffn_b1   = (const float*)d_in[25];
  const float* ffn_w2   = (const float*)d_in[26];
  const float* ffn_b2   = (const float*)d_in[27];
  const float* sg_w1    = (const float*)d_in[28];
  const float* sg_b1    = (const float*)d_in[29];
  const float* sg_dw    = (const float*)d_in[30];
  const float* sg_w2    = (const float*)d_in[31];
  const float* sg_b2    = (const float*)d_in[32];
  (void)in_sizes; (void)n_in; (void)out_size; (void)ws_size;

  float* out = (float*)d_out;
  float* ws  = (float*)d_ws;

  float* xer = ws;                  // x_eig / u_out (in-place scan)
  float* xei = ws + N32;
  float* GA  = ws + 2 * N32;        // xn (with GB) -> g1 -> spg -> dec_re
  float* GB  = ws + 3 * N32;        // xn hi       -> g2        -> dec_im
  float* SM  = ws + 4 * N32;        // small f32 region
  float* winv_re = SM;            float* winv_im = SM + 1024;
  float* xmr = SM + 2048;         float* xmi = SM + 3584;
  float* gate = SM + 5120;
  float* srcr = SM + 6656;        float* srci = SM + 8192;
  float* odr = SM + 9728;         float* odi = SM + 11264;
  float* ofr = SM + 12800;        float* ofi = SM + 14336;
  float* lnm = SM + 15872;        float* lnr = SM + 212480;
  _Float16* HB = (_Float16*)(SM + 409088);   // 32B aligned
  _Float16* pk_conv = HB;                    // 18*4*512 = 36864
  _Float16* pk_wbr  = HB + 36864;            // 1024
  _Float16* pk_wbi  = HB + 37888;
  _Float16* pk_wir  = HB + 38912;
  _Float16* pk_wii  = HB + 39936;
  _Float16* pk_f1   = HB + 40960;            // 8192
  _Float16* pk_f2   = HB + 49152;            // 8192
  _Float16* pk_sg1  = HB + 57344;            // 2048
  _Float16* pk_sg2  = HB + 59392;            // 1024

  // setup: inverse + weight packing
  k_inv<<<1, 32, 0, stream>>>(W_re, W_im, winv_re, winv_im);
  k_pack_rm<<<1, 64, 0, stream>>>(W_re, pk_wbr, 32, 32, 0);
  k_pack_rm<<<1, 64, 0, stream>>>(W_im, pk_wbi, 32, 32, 0);
  k_pack_rm<<<1, 64, 0, stream>>>(winv_re, pk_wir, 32, 32, 0);
  k_pack_rm<<<1, 64, 0, stream>>>(winv_im, pk_wii, 32, 32, 0);
  k_pack_rm<<<2, 256, 0, stream>>>(ffn_w1, pk_f1, 64, 128, 0);
  k_pack_rm<<<2, 256, 0, stream>>>(ffn_w2, pk_f2, 128, 64, 0);
  k_pack_rm<<<1, 128, 0, stream>>>(sg_w1, pk_sg1, 64, 32, 1);
  k_pack_rm<<<1, 64, 0, stream>>>(sg_w2, pk_sg2, 32, 32, 1);
  k_pack_conv<<<9, 256, 0, stream>>>(conv_w, pk_conv);

  // stage 1: LN + conv3x3 + residual (x_post -> d_out planes)
  k_ln_spatial<<<768, 256, 0, stream>>>(x_re, x_im, ln_s_g, ln_s_b, GA);
  k_conv<<<12288, 128, 0, stream>>>(GA, pk_conv, conv_b, x_re, x_im, out);

  // stage 2: encode to eigen basis
  k_encode<<<3072, 128, 0, stream>>>(out, pk_wbr, pk_wbi, xer, xei);

  // flux path
  k_xmean<<<1536, 256, 0, stream>>>(xer, xei, xmr, xmi);
  k_flux<<<1, 64, 0, stream>>>(dt_seq, lam_flux, bf_re, bf_im, flux_re, flux_im,
                               c_re, c_im, g_w, g_b, a_decay, b_freq, xmr, xmi,
                               gate, srcr, srci, odr, odi, ofr, ofi,
                               out + 2 * N32 + 524288);

  // spatial gate chain
  k_pw1<<<3072, 128, 0, stream>>>(xer, xei, pk_sg1, sg_b1, GA);
  k_dw<<<24576, 256, 0, stream>>>(GA, sg_dw, GB);
  k_pw2<<<3072, 128, 0, stream>>>(GB, pk_sg2, sg_b2, GA);

  // temporal scan (in place) + h_out
  k_scan<<<1024, 256, 0, stream>>>(xer, xei, GA, gate, srcr, srci, odr, odi,
                                   ofr, ofi, h_re, h_im, out + 2 * N32);

  // decode + LN stats + fused FFN / final residual
  k_decode<<<3072, 128, 0, stream>>>(xer, xei, pk_wir, pk_wii, GA, GB);
  k_lnstats<<<768, 256, 0, stream>>>(GA, GB, lnm, lnr);
  k_ffn<<<3072, 128, 0, stream>>>(GA, GB, lnm, lnr, ln_t_g, ln_t_b,
                                  pk_f1, ffn_b1, pk_f2, ffn_b2, out);
}
